// MultiHeadAttention_60284160966974
// MI455X (gfx1250) — matrix operand order repaired and checked by reading, not compiled, not run on hardware
//
#include <hip/hip_runtime.h>
#include <hip/hip_bf16.h>
#include <stdint.h>

#ifndef __has_builtin
#define __has_builtin(x) 0
#endif

// ---------------- problem constants ----------------
constexpr int Bz   = 8;
constexpr int Nn   = 1024;
constexpr int DEMB = 1024;
constexpr int Hh   = 8;
constexpr int DH   = 128;
constexpr int M1   = Bz * Nn;        // 8192 rows
constexpr int N1   = 4 * Hh * DH;    // 4096 qkvg cols
constexpr size_t SEC = (size_t)Bz * Hh * Nn * DH;  // 8388608 halves per section

typedef __attribute__((ext_vector_type(16))) _Float16 v16h;
typedef __attribute__((ext_vector_type(2)))  _Float16 v2h;
typedef __attribute__((ext_vector_type(8)))  float    v8f;

// ---------------- CDNA5 async global->LDS copy (ASYNCcnt-tracked) --------------
// Low 32 bits of a flat pointer to __shared__ memory are the DS-relative byte
// address (ISA 10.2 aperture rules), which is exactly what VDST wants.
__device__ static inline void async_ld16(void* lds_dst, const void* gsrc) {
  asm volatile("global_load_async_to_lds_b128 %0, %1, off"
               :: "v"((uint32_t)(uintptr_t)lds_dst), "v"(gsrc)
               : "memory");
}
__device__ static inline void wait_async0() {
#if __has_builtin(__builtin_amdgcn_s_wait_asynccnt)
  __builtin_amdgcn_s_wait_asynccnt(0);
#else
  asm volatile("s_wait_asynccnt 0x0" ::: "memory");
#endif
}

// ---------------- WMMA fragment-order swizzle helpers (ISA 7.12.2) -------------
// A 16x32 f16 tile, in this code's contiguous k order: lane l<16 holds row l, K {0..15}; lane l>=16 holds
// row l-16, K {16..31}. Within a lane, half-slot e maps to
// k = e + 16*(lane>=16). Inverse used by the stagers:
__device__ static inline void a_swz(int row, int k, int& lane, int& e) {
  lane = row + 16 * (k >> 4);
  e    = k & 15;
}
// B 32x16 f16 tile: element (k,n) -> lane = n + 16*(k>>4), slot e = k&15.
__device__ static inline void b_swz(int k, int n, int& lane, int& e) {
  lane = n + 16 * (k >> 4);
  e    = k & 15;
}

__device__ static inline v8f wmma_f16(v16h a, v16h b, v8f c) {
  return __builtin_amdgcn_wmma_f32_16x16x32_f16(false, a, false, b, (short)0, c,
                                                false, false);
}

// B fragment straight from row-major f16 source: element (k,n) = base[k*kstride+n]
__device__ static inline v16h frag_b_k(const _Float16* base, int kstride) {
  const int lane  = threadIdx.x & 31;
  const int n     = lane & 15;
  const int kbase = (lane >> 4) * 16;
  v16h b;
#pragma unroll
  for (int j = 0; j < 8; ++j) {
    b[2 * j]     = base[(kbase + 2 * j) * kstride + n];
    b[2 * j + 1] = base[(kbase + 2 * j + 1) * kstride + n];
  }
  return b;
}

// B fragment from transposed f16 source (k contiguous): element (k,n) = base[n*nstride+k]
__device__ static inline v16h frag_bT(const _Float16* base, int nstride) {
  const int lane  = threadIdx.x & 31;
  const int n     = lane & 15;
  const int kbase = (lane >> 4) * 16;
  v16h b;
#pragma unroll
  for (int j = 0; j < 8; ++j) {
    const v2h p = *(const v2h*)(base + n * nstride + kbase + 2 * j);
    b[2 * j]     = p[0];
    b[2 * j + 1] = p[1];
  }
  return b;
}

// A fragment gather from row-major f16 LDS (used for P in attention phase C)
__device__ static inline v16h frag_a_rm(const _Float16* base, int row_stride) {
  const int lane = threadIdx.x & 31;
  const int row  = lane & 15;
  const int koff = (lane >> 4) * 16;
  v16h a;
#pragma unroll
  for (int j = 0; j < 8; ++j) {
    const int k = 2 * j + koff;
    a[2 * j]     = base[row * row_stride + k];
    a[2 * j + 1] = base[row * row_stride + k + 1];
  }
  return a;
}

// ---------------- kernel 1: qkvg = X @ W_att, scatter into Q/K/V/G -------------
#define TM 128
#define TN 128
#define TK 32

__global__ __launch_bounds__(256) void k_qkvg(const float* __restrict__ X,
                                              const float* __restrict__ W,
                                              _Float16* __restrict__ ws) {
  // fragment-order tiles: [16x16-tile index][lane][16 half slots]
  __shared__ __align__(32) _Float16 As[TM / 16][32][16];  // 8 KB
  __shared__ __align__(32) _Float16 Bs[TN / 16][32][16];  // 8 KB
  const int tid  = threadIdx.x;
  const int bm   = blockIdx.y * TM;
  const int bn   = blockIdx.x * TN;
  const int wave = tid >> 5, lane = tid & 31;
  const int wm = (wave >> 1) * 32, wn = (wave & 1) * 64;

  v8f acc[2][4] = {};

  for (int kt = 0; kt < DEMB / TK; ++kt) {
    // stage A: 128x32 fp32 -> f16 swizzled (1024 float4 loads)
#pragma unroll
    for (int i = 0; i < 4; ++i) {
      const int slot = tid + i * 256;
      const int r = slot >> 3, k0 = (slot & 7) * 4;
      const float4 v =
          ((const float4*)(X + (size_t)(bm + r) * DEMB + kt * TK))[slot & 7];
      int ln, e;
      a_swz(r & 15, k0, ln, e);
      _Float16* d = &As[r >> 4][ln][e];
      d[0] = (_Float16)v.x; d[1] = (_Float16)v.y;
      d[2] = (_Float16)v.z; d[3] = (_Float16)v.w;
    }
    // stage B: 32x128 fp32 -> f16 swizzled
#pragma unroll
    for (int i = 0; i < 4; ++i) {
      const int slot = tid + i * 256;
      const int kr = slot >> 5, n0 = (slot & 31) * 4;
      const float4 v =
          ((const float4*)(W + (size_t)(kt * TK + kr) * N1 + bn))[slot & 31];
      const int t = n0 >> 4;
      int ln, e;
      b_swz(kr, n0 & 15, ln, e);
      Bs[t][ln + 0][e] = (_Float16)v.x;
      Bs[t][ln + 1][e] = (_Float16)v.y;
      Bs[t][ln + 2][e] = (_Float16)v.z;
      Bs[t][ln + 3][e] = (_Float16)v.w;
    }
    __syncthreads();

    v16h af[2], bf[4];
#pragma unroll
    for (int mi = 0; mi < 2; ++mi) af[mi] = *(const v16h*)&As[(wm >> 4) + mi][lane][0];
#pragma unroll
    for (int ni = 0; ni < 4; ++ni) bf[ni] = *(const v16h*)&Bs[(wn >> 4) + ni][lane][0];
#pragma unroll
    for (int mi = 0; mi < 2; ++mi)
#pragma unroll
      for (int ni = 0; ni < 4; ++ni) acc[mi][ni] = wmma_f16(af[mi], bf[ni], acc[mi][ni]);
    __syncthreads();
  }

  // epilogue: scatter into Q/K/V/G sections laid out [B,H,N,128]
  const int rbase = 8 * (lane >> 4);
  const int cl    = lane & 15;
#pragma unroll
  for (int mi = 0; mi < 2; ++mi)
#pragma unroll
    for (int ni = 0; ni < 4; ++ni) {
      const int gc   = bn + wn + ni * 16 + cl;
      const int sect = gc >> 10;
      const int h    = (gc >> 7) & 7;
      const int d    = gc & 127;
      _Float16* dst  = ws + (size_t)sect * SEC;
#pragma unroll
      for (int r = 0; r < 8; ++r) {
        const int gm = bm + wm + mi * 16 + rbase + r;
        const int b  = gm >> 10, n = gm & 1023;
        dst[(((size_t)(b * Hh + h)) * Nn + n) * DH + d] = (_Float16)acc[mi][ni][r];
      }
    }
}

// ---------------- kernel 2: attention per (b,h,16-query block) -----------------
__global__ __launch_bounds__(256) void k_attn(const _Float16* __restrict__ ws,
                                              const float* __restrict__ bias,
                                              const uint8_t* __restrict__ mask,
                                              const float* __restrict__ gamma,
                                              _Float16* __restrict__ att) {
  const _Float16* Q = ws;
  const _Float16* K = ws + SEC;
  const _Float16* V = ws + 2 * SEC;
  const _Float16* G = ws + 3 * SEC;

  __shared__ __align__(32) _Float16 qs[4][32][16];  // Q tile in A-frag order, 4 KB
  __shared__ _Float16 sc[16][Nn];                   // scores/probs f16, 32 KB
  __shared__ __align__(16) _Float16 vt[32][DH];     // V k-slice, 8 KB
  __shared__ float red[16][17];

  const int bh = blockIdx.y;           // b*H + h
  const int b = bh >> 3, h = bh & 7;
  const int q0 = blockIdx.x * 16;
  const int tid = threadIdx.x, wave = tid >> 5, lane = tid & 31;

  // async-stage Q tile (16x128 halves = 256 x 16B) into swizzled A layout
  {
    const _Float16* qsrc = Q + ((size_t)bh * Nn + q0) * DH;
    const int row = tid >> 4, f0 = (tid & 15) * 8;
    int ln, e;
    a_swz(row, f0 & 31, ln, e);
    async_ld16(&qs[f0 >> 5][ln][e], qsrc + (size_t)row * DH + f0);
    wait_async0();
  }
  __syncthreads();

  const float scale = 0.08838834764831845f;  // 1/sqrt(128)
  const float gm_h  = gamma[h];

  // Phase A: S = Q K^T * scale + gamma*bias, masked -> sc (f16)
  v16h qf[4];
#pragma unroll
  for (int ks = 0; ks < 4; ++ks) qf[ks] = *(const v16h*)&qs[ks][lane][0];

  for (int i = 0; i < 8; ++i) {
    const int kt = wave * 8 + i;
    const int k0 = kt * 16;
    v8f s = {};
#pragma unroll
    for (int ks = 0; ks < 4; ++ks) {
      const v16h bf = frag_bT(K + ((size_t)bh * Nn + k0) * DH + ks * 32, DH);
      s = wmma_f16(qf[ks], bf, s);
    }
    const int rq = 8 * (lane >> 4), n = lane & 15;
#pragma unroll
    for (int r = 0; r < 8; ++r) {
      const int qr = q0 + rq + r;
      const int kc = k0 + n;
      const size_t ix = ((size_t)b * Nn + qr) * Nn + kc;
      float v = s[r] * scale + gm_h * bias[ix];
      if (mask[ix]) v = -30000.0f;     // f16-safe "-inf"
      sc[rq + r][kc] = (_Float16)v;
    }
  }
  __syncthreads();

  // Phase B: row softmax (16 threads per row)
  {
    const int row = tid >> 4, sub = tid & 15;
    float m = -3.4e38f;
    for (int c = sub; c < Nn; c += 16) m = fmaxf(m, (float)sc[row][c]);
    red[row][sub] = m;
    __syncthreads();
    if (sub == 0) {
      float mm = red[row][0];
      for (int j = 1; j < 16; ++j) mm = fmaxf(mm, red[row][j]);
      red[row][16] = mm;
    }
    __syncthreads();
    const float mm = red[row][16];
    float ssum = 0.f;
    for (int c = sub; c < Nn; c += 16) ssum += __expf((float)sc[row][c] - mm);
    red[row][sub] = ssum;
    __syncthreads();
    if (sub == 0) {
      float t = 0.f;
      for (int j = 0; j < 16; ++j) t += red[row][j];
      red[row][16] = 1.0f / t;
    }
    __syncthreads();
    const float inv = red[row][16];
    for (int c = sub; c < Nn; c += 16)
      sc[row][c] = (_Float16)(__expf((float)sc[row][c] - mm) * inv);
  }
  __syncthreads();

  // Phase C: O = P V. V k-slice (32x128, contiguous 8KB) async-staged to LDS,
  // shared by all 8 waves; each wave owns 16 output features.
  {
    const int f0 = wave * 16;
    v8f o = {};
    for (int kk = 0; kk < Nn / 32; ++kk) {
      const char* vsrc = (const char*)(V + ((size_t)bh * Nn + kk * 32) * DH);
      char* vdst = (char*)&vt[0][0];
      async_ld16(vdst + tid * 16,        vsrc + tid * 16);
      async_ld16(vdst + (tid + 256) * 16, vsrc + (tid + 256) * 16);
      wait_async0();
      __syncthreads();
      const v16h a  = frag_a_rm(&sc[0][kk * 32], Nn);
      const v16h bf = frag_b_k(&vt[0][f0], DH);
      o = wmma_f16(a, bf, o);
      __syncthreads();
    }
    const int rq = 8 * (lane >> 4), n = lane & 15;
#pragma unroll
    for (int r = 0; r < 8; ++r) {
      const int qr = q0 + rq + r;
      const float g = (float)G[((size_t)bh * Nn + qr) * DH + f0 + n];
      const float val = o[r] * (1.0f / (1.0f + __expf(-g)));
      att[((size_t)(b * Nn + qr)) * (Hh * DH) + h * DH + f0 + n] = (_Float16)val;
    }
  }
}

// ---------------- kernel 3: h = att @ W_ff + b_ff + x  -> d_out (fp32) ---------
__global__ __launch_bounds__(256) void k_ff(const _Float16* __restrict__ A,
                                            const float* __restrict__ W,
                                            const float* __restrict__ bff,
                                            const float* __restrict__ X,
                                            float* __restrict__ out) {
  __shared__ __align__(32) _Float16 As[TM / 16][32][16];  // 8 KB
  __shared__ __align__(32) _Float16 Bs[TN / 16][32][16];  // 8 KB
  const int tid  = threadIdx.x;
  const int bm   = blockIdx.y * TM;
  const int bn   = blockIdx.x * TN;
  const int wave = tid >> 5, lane = tid & 31;
  const int wm = (wave >> 1) * 32, wn = (wave & 1) * 64;

  v8f acc[2][4] = {};

  for (int kt = 0; kt < DEMB / TK; ++kt) {
    // stage A: f16 tile via async 16B copies straight into swizzled layout
    // (8 consecutive aligned K land in one lane with consecutive slots)
#pragma unroll
    for (int i = 0; i < 2; ++i) {
      const int slot = tid + i * 256;           // 512 x 16B
      const int r = slot >> 2, k0 = (slot & 3) * 8;
      int ln, e;
      a_swz(r & 15, k0, ln, e);
      async_ld16(&As[r >> 4][ln][e],
                 A + (size_t)(bm + r) * DEMB + kt * TK + k0);
    }
    // stage B: 32x128 fp32 -> f16 swizzled
#pragma unroll
    for (int i = 0; i < 4; ++i) {
      const int slot = tid + i * 256;
      const int kr = slot >> 5, n0 = (slot & 31) * 4;
      const float4 v =
          ((const float4*)(W + (size_t)(kt * TK + kr) * DEMB + bn))[slot & 31];
      const int t = n0 >> 4;
      int ln, e;
      b_swz(kr, n0 & 15, ln, e);
      Bs[t][ln + 0][e] = (_Float16)v.x;
      Bs[t][ln + 1][e] = (_Float16)v.y;
      Bs[t][ln + 2][e] = (_Float16)v.z;
      Bs[t][ln + 3][e] = (_Float16)v.w;
    }
    wait_async0();
    __syncthreads();

    v16h af[2], bf[4];
#pragma unroll
    for (int mi = 0; mi < 2; ++mi) af[mi] = *(const v16h*)&As[(wm >> 4) + mi][lane][0];
#pragma unroll
    for (int ni = 0; ni < 4; ++ni) bf[ni] = *(const v16h*)&Bs[(wn >> 4) + ni][lane][0];
#pragma unroll
    for (int mi = 0; mi < 2; ++mi)
#pragma unroll
      for (int ni = 0; ni < 4; ++ni) acc[mi][ni] = wmma_f16(af[mi], bf[ni], acc[mi][ni]);
    __syncthreads();
  }

  const int rbase = 8 * (lane >> 4);
  const int cl    = lane & 15;
#pragma unroll
  for (int mi = 0; mi < 2; ++mi)
#pragma unroll
    for (int ni = 0; ni < 4; ++ni) {
      const int gc = bn + wn + ni * 16 + cl;
#pragma unroll
      for (int r = 0; r < 8; ++r) {
        const int gm = bm + wm + mi * 16 + rbase + r;
        const size_t ix = (size_t)gm * DEMB + gc;
        out[ix] = acc[mi][ni][r] + bff[gc] + X[ix];
      }
    }
}

// ---------------- kernel 4: in-place LayerNorm over last dim -------------------
__global__ __launch_bounds__(256) void k_ln(float* __restrict__ out,
                                            const float* __restrict__ g,
                                            const float* __restrict__ bta) {
  const int tid = threadIdx.x;
  float* p = out + (size_t)blockIdx.x * DEMB;
  __shared__ float red[256];

  float loc[4], s = 0.f;
#pragma unroll
  for (int i = 0; i < 4; ++i) { loc[i] = p[tid + i * 256]; s += loc[i]; }
  red[tid] = s; __syncthreads();
  for (int off = 128; off > 0; off >>= 1) {
    if (tid < off) red[tid] += red[tid + off];
    __syncthreads();
  }
  const float mu = red[0] * (1.0f / DEMB);
  __syncthreads();

  float vs = 0.f;
#pragma unroll
  for (int i = 0; i < 4; ++i) { const float d = loc[i] - mu; vs += d * d; }
  red[tid] = vs; __syncthreads();
  for (int off = 128; off > 0; off >>= 1) {
    if (tid < off) red[tid] += red[tid + off];
    __syncthreads();
  }
  const float inv = rsqrtf(red[0] * (1.0f / DEMB) + 1e-5f);
#pragma unroll
  for (int i = 0; i < 4; ++i) {
    const int c = tid + i * 256;
    p[c] = (loc[i] - mu) * inv * g[c] + bta[c];
  }
}

// ---------------- host entry ---------------------------------------------------
extern "C" void kernel_launch(void* const* d_in, const int* in_sizes, int n_in,
                              void* d_out, int out_size, void* d_ws, size_t ws_size,
                              hipStream_t stream) {
  const float*   x     = (const float*)d_in[0];
  const uint8_t* mask  = (const uint8_t*)d_in[1];  // jax bool -> 1 byte
  const float*   bias  = (const float*)d_in[2];
  const float*   gamma = (const float*)d_in[3];
  const float*   W_att = (const float*)d_in[4];
  const float*   W_ff  = (const float*)d_in[5];
  const float*   b_ff  = (const float*)d_in[6];
  const float*   ln_g  = (const float*)d_in[7];
  const float*   ln_b  = (const float*)d_in[8];
  float* out = (float*)d_out;

  _Float16* ws  = (_Float16*)d_ws;   // needs 5*SEC halves = 80 MB
  _Float16* att = ws + 4 * SEC;

  dim3 g1(N1 / TN, M1 / TM);         // 32 x 64
  k_qkvg<<<g1, 256, 0, stream>>>(x, W_att, ws);

  dim3 g2(Nn / 16, Bz * Hh);         // 64 x 64
  k_attn<<<g2, 256, 0, stream>>>(ws, bias, mask, gamma, att);

  dim3 g3(DEMB / TN, M1 / TM);       // 8 x 64
  k_ff<<<g3, 256, 0, stream>>>(att, W_ff, b_ff, x, out);

  k_ln<<<M1, 256, 0, stream>>>(out, ln_g, ln_b);
}
